// RobertaHybridEncoder_81363860456153
// MI455X (gfx1250) — matrix operand order repaired and checked by reading, not compiled, not run on hardware
//
#include <hip/hip_runtime.h>

// ---------------------------------------------------------------------------
// CDNA5 (gfx1250) types + WMMA helper
// ---------------------------------------------------------------------------
typedef _Float16 v16h __attribute__((ext_vector_type(16)));
typedef _Float16 v8h  __attribute__((ext_vector_type(8)));
typedef float    v8f  __attribute__((ext_vector_type(8)));

#define WMMA_F16(a, b, c) \
  __builtin_amdgcn_wmma_f32_16x16x32_f16(false, (a), false, (b), (short)0, (c), false, false)

// A fragment (16x32 f16, M rows x K): per ISA table:
//   lanes 0-15 : halves 0..7 = K k+0..7 , halves 8..15 = K k+16..23
//   lanes16-31 : halves 0..7 = K k+8..15, halves 8..15 = K k+24..31
__device__ __forceinline__ v16h load_a_frag(const _Float16* __restrict__ A, int lda,
                                            long row, int k, int laneHi) {
  const _Float16* p = A + row * (long)lda + k + (laneHi ? 8 : 0);
  v8h lo = *(const v8h*)(p);
  v8h hi = *(const v8h*)(p + 16);
  v16h r;
#pragma unroll
  for (int i = 0; i < 8; i++) { r[i] = lo[i]; r[8 + i] = hi[i]; }
  return r;
}

// B fragment (32x16 f16, K x N cols): same K layout as A fragment:
// lanes 0-15 hold K k+0..7, k+16..23; lanes 16-31 hold K k+8..15, k+24..31.
// Works for both global and LDS-resident weights (compiler picks vmem vs ds path).
__device__ __forceinline__ v16h load_bw_frag(const _Float16* W, int ldw,
                                             int n, int k, int laneHi) {
  const _Float16* p = W + (long)n * ldw + k + (laneHi ? 8 : 0);
  v8h lo = *(const v8h*)(p);
  v8h hi = *(const v8h*)(p + 16);
  v16h r;
#pragma unroll
  for (int i = 0; i < 8; i++) { r[i] = lo[i]; r[8 + i] = hi[i]; }
  return r;
}

// ---------------------------------------------------------------------------
// Generic batched f16 WMMA GEMM:  Out = scale * ( act( A @ Bw^T + bias ) )
//   A:[M x K] row-major f16.  Bw: [N x K] row-major f16 (weight layout).
//   Block = 256 thr = 8 waves; wave = 16M x 64N tile.
// ---------------------------------------------------------------------------
__global__ __launch_bounds__(256) void gemm_h_kernel(
    const _Float16* __restrict__ A, long sA, int lda,
    const _Float16* __restrict__ Bw, long sB, int ldb,
    float* __restrict__ OutF, _Float16* __restrict__ OutH, long sO, int ldo,
    const float* __restrict__ bias, int relu, const float* __restrict__ scale_ptr,
    int M, int N, int K) {
  const int bz = blockIdx.z;
  const _Float16* Ab = A + (long)bz * sA;
  const _Float16* Bb = Bw + (long)bz * sB;
  const int w = threadIdx.x >> 5, lane = threadIdx.x & 31;
  const int laneHi = lane >> 4, ln = lane & 15;
  const int mTile = (blockIdx.y * 8 + w) * 16;
  const int nBase = blockIdx.x * 64;
  if (mTile >= M) return;  // wave-uniform
  const long row = mTile + ln;

  v8f acc[4] = {};
  for (int k = 0; k < K; k += 32) {
    // Prefetch next k-slice of this row into WGP caches (global_prefetch_b8).
    if (k + 128 < K) __builtin_prefetch(A + row * (long)lda + k + 128, 0, 1);
    v16h a = load_a_frag(Ab, lda, row, k, laneHi);
#pragma unroll
    for (int j = 0; j < 4; j++) {
      int n = nBase + j * 16 + ln;
      v16h bf = {};
      if (n < N) bf = load_bw_frag(Bb, ldb, n, k, laneHi);
      acc[j] = WMMA_F16(a, bf, acc[j]);
    }
  }

  const float scale = scale_ptr ? scale_ptr[0] : 1.0f;
#pragma unroll
  for (int j = 0; j < 4; j++) {
    int n = nBase + j * 16 + ln;
    if (n >= N) continue;
    float bv = bias ? bias[n] : 0.0f;
#pragma unroll
    for (int r = 0; r < 8; r++) {
      long rr = mTile + r + laneHi * 8;  // C/D: VGPR index = row, lane = col
      float v = acc[j][r] + bv;
      if (relu) v = v > 0.0f ? v : 0.0f;
      v *= scale;
      long idx = (long)bz * sO + rr * (long)ldo + n;
      if (OutF) OutF[idx] = v;
      if (OutH) OutH[idx] = (_Float16)v;
    }
  }
}

// ---------------------------------------------------------------------------
// Fused pairwise-MLP logits kernel (the dominant stage).
// pre[m; s,t] = Arow[s,m] + Brow[t,m] + W1cd @ [Zj_s*Zi_t ; |Zj_s-Zi_t|]  (K=48->64)
// logit[s,t]  = sum_m relu(pre)*W2[m] + b2 + mask
// Grid: (S/16 s-tiles, S/128, B); each wave owns 16 t-rows, loops 16 s values.
// W1cd (12 KB) is staged once per workgroup into LDS via the CDNA5 async
// global->LDS DMA path (ASYNCcnt + s_wait_asynccnt), then all 8 waves build
// their resident B fragments from LDS (ds_load) instead of 8x global re-reads.
// ---------------------------------------------------------------------------
__global__ __launch_bounds__(256) void pair_logits_kernel(
    const float* __restrict__ Zj, const float* __restrict__ Zi,    // [B*S][32] f32
    const float* __restrict__ Arow, const float* __restrict__ Brow,// [B*S][96] f32
    const _Float16* __restrict__ W1cd,                             // [96][64] f16
    const float* __restrict__ W2, const float* __restrict__ b2,
    const float* __restrict__ mask,                                // [B*S]
    float* __restrict__ logits, int S) {
  __shared__ _Float16 w1cd_lds[96 * 64];

#if __has_builtin(__builtin_amdgcn_global_load_async_to_lds_b128)
  {
    // 12288 bytes = 768 chunks of 16B; 256 threads x 3 chunks, all lanes active.
    typedef int v4i_ld __attribute__((vector_size(16)));
    typedef __attribute__((address_space(1))) v4i_ld* gvec_p;
    typedef __attribute__((address_space(3))) v4i_ld* lvec_p;
    gvec_p g = (gvec_p)W1cd;
    lvec_p l = (lvec_p)w1cd_lds;
    for (int c = threadIdx.x; c < (96 * 64 * 2) / 16; c += 256)
      __builtin_amdgcn_global_load_async_to_lds_b128(g + c, l + c, 0, 0);
#if __has_builtin(__builtin_amdgcn_s_wait_asynccnt)
    __builtin_amdgcn_s_wait_asynccnt(0);
#else
    asm volatile("s_wait_asynccnt 0x0" ::: "memory");
#endif
  }
#else
  for (int c = threadIdx.x; c < 96 * 64; c += 256) w1cd_lds[c] = W1cd[c];
#endif
  __syncthreads();

  const int bz = blockIdx.z;
  const int sBase = blockIdx.x * 16;
  const int w = threadIdx.x >> 5, lane = threadIdx.x & 31;
  const int laneHi = lane >> 4, ln = lane & 15;
  const int tBase = blockIdx.y * 128 + w * 16;

  // Resident B fragments of W1cd^T from LDS: 6 N-tiles x 2 k-steps (K = 0..63).
  v16h bf[6][2];
#pragma unroll
  for (int j = 0; j < 6; j++)
#pragma unroll
    for (int kk = 0; kk < 2; kk++)
      bf[j][kk] = load_bw_frag(w1cd_lds, 64, j * 16 + ln, kk * 32, laneHi);

  // This lane's Zi row (A-matrix row t = tBase + ln).
  float zi[24];
  {
    const float* zp = Zi + ((long)bz * S + tBase + ln) * 32;
#pragma unroll
    for (int d = 0; d < 24; d++) zi[d] = zp[d];
  }
  // Epilogue constants: Brow[t, n] for this lane's (row, col) cells, W2[n].
  float brv[6][8], w2v[6];
#pragma unroll
  for (int r = 0; r < 8; r++) {
    long t = tBase + r + laneHi * 8;
#pragma unroll
    for (int j = 0; j < 6; j++)
      brv[j][r] = Brow[((long)bz * S + t) * 96 + j * 16 + ln];
  }
#pragma unroll
  for (int j = 0; j < 6; j++) w2v[j] = W2[j * 16 + ln];
  const float b2v = b2[0];
  const float FMINV = -3.402823466e38f;

  for (int s = 0; s < 16; s++) {
    const long srow = (long)bz * S + sBase + s;
    const float* sp = Zj + srow * 32;
    if (s < 15) __builtin_prefetch(sp + 32, 0, 1);  // next Zj row
    float prod[24], ad[24];
#pragma unroll
    for (int d = 0; d < 24; d++) {
      float zj = sp[d];
      prod[d] = zj * zi[d];
      ad[d] = fabsf(zj - zi[d]);
    }
    // Build A fragments (features K: 0..23 = prod, 24..47 = |diff|, 48..63 = 0)
    // honoring the interleaved per-half-wave K layout.
    v16h a0, a1;
#define FEAT(K) ((K) < 24 ? prod[(K)] : ((K) < 48 ? ad[(K) - 24] : 0.0f))
    if (!laneHi) {
#pragma unroll
      for (int i = 0; i < 8; i++) {
        a0[i] = (_Float16)FEAT(i);          a0[8 + i] = (_Float16)FEAT(16 + i);
        a1[i] = (_Float16)FEAT(32 + i);     a1[8 + i] = (_Float16)FEAT(48 + i);
      }
    } else {
#pragma unroll
      for (int i = 0; i < 8; i++) {
        a0[i] = (_Float16)FEAT(8 + i);      a0[8 + i] = (_Float16)FEAT(24 + i);
        a1[i] = (_Float16)FEAT(40 + i);     a1[8 + i] = (_Float16)FEAT(56 + i);
      }
    }
#undef FEAT

    float plog[8] = {};
#pragma unroll
    for (int j = 0; j < 6; j++) {
      v8f acc = {};
      acc = WMMA_F16(a0, bf[j][0], acc);
      acc = WMMA_F16(a1, bf[j][1], acc);
      float ar = Arow[srow * 96 + j * 16 + ln];  // includes b1
#pragma unroll
      for (int r = 0; r < 8; r++) {
        float pre = acc[r] + ar + brv[j][r];
        pre = pre > 0.0f ? pre : 0.0f;  // ReLU
        plog[r] += pre * w2v[j];
      }
    }
    // Reduce over the 16 output columns held across lanes of each half-wave.
#pragma unroll
    for (int r = 0; r < 8; r++)
#pragma unroll
      for (int m = 1; m < 16; m <<= 1) plog[r] += __shfl_xor(plog[r], m, 32);

    if (ln == 0) {
      int t0 = tBase + laneHi * 8;
#pragma unroll
      for (int r = 0; r < 8; r++) {
        long t = t0 + r;
        float lg = plog[r] + b2v + (1.0f - mask[(long)bz * S + t]) * FMINV;
        logits[srow * S + t] = lg;
      }
    }
  }
}

// ---------------------------------------------------------------------------
// Row softmax over t (S=512) -> f16 probs.  One block per (b,s).
// ---------------------------------------------------------------------------
__global__ __launch_bounds__(256) void softmax_kernel(const float* __restrict__ lg,
                                                      _Float16* __restrict__ ph, int S) {
  __shared__ float red[256];
  const long row = blockIdx.x;
  const float* lp = lg + row * S;
  const int tid = threadIdx.x;
  float mx = -3.4e38f;
  for (int t = tid; t < S; t += 256) mx = fmaxf(mx, lp[t]);
  red[tid] = mx; __syncthreads();
  for (int s2 = 128; s2 > 0; s2 >>= 1) {
    if (tid < s2) red[tid] = fmaxf(red[tid], red[tid + s2]);
    __syncthreads();
  }
  mx = red[0]; __syncthreads();
  float sum = 0.0f;
  for (int t = tid; t < S; t += 256) sum += __expf(lp[t] - mx);
  red[tid] = sum; __syncthreads();
  for (int s2 = 128; s2 > 0; s2 >>= 1) {
    if (tid < s2) red[tid] += red[tid + s2];
    __syncthreads();
  }
  const float inv = 1.0f / red[0];
  for (int t = tid; t < S; t += 256) ph[row * S + t] = (_Float16)(__expf(lp[t] - mx) * inv);
}

// ---------------------------------------------------------------------------
// Small elementwise helpers
// ---------------------------------------------------------------------------
__global__ void cast_h_kernel(const float* __restrict__ s, _Float16* __restrict__ d, long n) {
  long i = (long)blockIdx.x * blockDim.x + threadIdx.x;
  if (i < n) d[i] = (_Float16)s[i];
}

// Transpose-cast per batch: src f32 [BATCH][R][C] -> dst f16 [BATCH][C][R].
// (Puts H_i into weight layout so the ctx GEMM uses contiguous B fragments.)
__global__ void transpose_h_kernel(const float* __restrict__ src, _Float16* __restrict__ dst,
                                   int R, int C) {
  long i = (long)blockIdx.x * blockDim.x + threadIdx.x;
  long per = (long)R * C;
  long b = i / per, rc = i % per;
  int r = (int)(rc / C), c = (int)(rc % C);
  dst[b * per + (long)c * R + r] = (_Float16)src[i];
}

// Weight [N x Kin] (with row stride srcLd, column offset colOfs) -> f16 [N x Kp], zero-padded K.
__global__ void wcvt_kernel(const float* __restrict__ src, int srcLd, int colOfs,
                            int N, int Kin, _Float16* __restrict__ dst, int Kp) {
  long i = (long)blockIdx.x * blockDim.x + threadIdx.x;
  if (i >= (long)N * Kp) return;
  int n = (int)(i / Kp), k = (int)(i % Kp);
  dst[i] = (_Float16)(k < Kin ? src[(long)n * srcLd + colOfs + k] : 0.0f);
}

// W1cd[96][64]: k<24 -> W1[m, 48+k] (prod block); k<48 -> W1[m, 72+(k-24)] (|diff| block); else 0.
__global__ void w1cd_kernel(const float* __restrict__ W1, _Float16* __restrict__ dst) {
  int i = blockIdx.x * blockDim.x + threadIdx.x;
  if (i >= 96 * 64) return;
  int m = i / 64, k = i % 64;
  float v = 0.0f;
  if (k < 24) v = W1[m * 96 + 48 + k];
  else if (k < 48) v = W1[m * 96 + 72 + (k - 24)];
  dst[i] = (_Float16)v;
}

// msg_in = [ctx, H_j, ctx*H_j] -> f16 [B*S x 3H]
__global__ void msgin_kernel(const float* __restrict__ ctx, const float* __restrict__ Hj,
                             _Float16* __restrict__ mi, long rows, int H) {
  long i = (long)blockIdx.x * blockDim.x + threadIdx.x;
  if (i >= rows * H) return;
  long row = i / H; int h = (int)(i % H);
  float c = ctx[i], x = Hj[i];
  _Float16* mp = mi + row * (long)(3 * H);
  mp[h] = (_Float16)c;
  mp[H + h] = (_Float16)x;
  mp[2 * H + h] = (_Float16)(c * x);
}

// ---------------------------------------------------------------------------
// Launcher
// ---------------------------------------------------------------------------
extern "C" void kernel_launch(void* const* d_in, const int* in_sizes, int n_in,
                              void* d_out, int out_size, void* d_ws, size_t ws_size,
                              hipStream_t stream) {
  (void)in_sizes; (void)n_in; (void)out_size; (void)ws_size;
  // setup_inputs() order (14 entries):
  const float* H_j  = (const float*)d_in[0];
  const float* H_i  = (const float*)d_in[1];
  const float* mask = (const float*)d_in[2];
  const float* Wpj  = (const float*)d_in[3];
  const float* Wpi  = (const float*)d_in[4];
  const float* W1   = (const float*)d_in[5];
  const float* b1   = (const float*)d_in[6];
  const float* W2   = (const float*)d_in[7];
  const float* b2   = (const float*)d_in[8];
  const float* Wv1  = (const float*)d_in[9];
  const float* bv1  = (const float*)d_in[10];
  const float* Wv2  = (const float*)d_in[11];
  const float* bv2  = (const float*)d_in[12];
  const float* alpha= (const float*)d_in[13];

  constexpr int B = 4, S = 512, H = 768, MH = 96;
  const long BS = (long)B * S;  // 2048

  char* p = (char*)d_ws;
  auto carve = [&](size_t bytes) -> char* {
    char* r = p;
    p += (bytes + 255) & ~(size_t)255;
    return r;
  };
  _Float16* Hj_h   = (_Float16*)carve(BS * H * 2);
  _Float16* HiT_h  = (_Float16*)carve(BS * H * 2);  // [B][H][S] transposed
  _Float16* Wpj_h  = (_Float16*)carve(24 * 768 * 2);
  _Float16* Wpi_h  = (_Float16*)carve(24 * 768 * 2);
  _Float16* W1a_h  = (_Float16*)carve(96 * 32 * 2);
  _Float16* W1b_h  = (_Float16*)carve(96 * 32 * 2);
  _Float16* W1cd_h = (_Float16*)carve(96 * 64 * 2);
  _Float16* Wv1_h  = (_Float16*)carve(768L * 2304 * 2);
  _Float16* Wv2_h  = (_Float16*)carve(768L * 768 * 2);
  _Float16* Hi_h   = (_Float16*)carve(BS * H * 2);  // row-major (for Zi projection A)
  float*    Zj     = (float*)   carve(BS * 32 * 4);
  float*    Zi     = (float*)   carve(BS * 32 * 4);
  _Float16* Zj_h   = (_Float16*)carve(BS * 32 * 2);
  _Float16* Zi_h   = (_Float16*)carve(BS * 32 * 2);
  float*    Arw    = (float*)   carve(BS * 96 * 4);
  float*    Brw    = (float*)   carve(BS * 96 * 4);
  float*    lgbuf  = (float*)   carve(BS * S * 4);
  _Float16* probs_h= (_Float16*)carve(BS * S * 2);
  float*    ctx    = (float*)   carve(BS * H * 4);
  _Float16* msgin  = (_Float16*)carve(BS * 2304L * 2);
  _Float16* hid2   = (_Float16*)carve(BS * H * 2);
  float* outF = (float*)d_out;

  const int T = 256;
  auto g1 = [](long n) { return (unsigned)((n + 255) / 256); };

  // --- converts / weight pads ---
  cast_h_kernel<<<g1(BS * H), T, 0, stream>>>(H_j, Hj_h, BS * H);
  cast_h_kernel<<<g1(BS * H), T, 0, stream>>>(H_i, Hi_h, BS * H);
  transpose_h_kernel<<<g1(BS * H), T, 0, stream>>>(H_i, HiT_h, S, H);  // per-b [S][H]->[H][S]
  wcvt_kernel<<<g1(24L * 768), T, 0, stream>>>(Wpj, 768, 0, 24, 768, Wpj_h, 768);
  wcvt_kernel<<<g1(24L * 768), T, 0, stream>>>(Wpi, 768, 0, 24, 768, Wpi_h, 768);
  wcvt_kernel<<<g1(96L * 32), T, 0, stream>>>(W1, 96, 0, 96, 24, W1a_h, 32);
  wcvt_kernel<<<g1(96L * 32), T, 0, stream>>>(W1, 96, 24, 96, 24, W1b_h, 32);
  w1cd_kernel<<<g1(96L * 64), T, 0, stream>>>(W1, W1cd_h);
  wcvt_kernel<<<g1(768L * 2304), T, 0, stream>>>(Wv1, 2304, 0, 768, 2304, Wv1_h, 2304);
  wcvt_kernel<<<g1(768L * 768), T, 0, stream>>>(Wv2, 768, 0, 768, 768, Wv2_h, 768);
  // Zero pad columns 24..31 of Z buffers (read as K=32 later).
  (void)hipMemsetAsync(Zj, 0, BS * 32 * 4, stream);
  (void)hipMemsetAsync(Zi, 0, BS * 32 * 4, stream);
  (void)hipMemsetAsync(Zj_h, 0, BS * 32 * 2, stream);
  (void)hipMemsetAsync(Zi_h, 0, BS * 32 * 2, stream);

  // --- projections: Z = H @ Wp^T  (M=2048, N=24, K=768) ---
  gemm_h_kernel<<<dim3(1, 16, 1), T, 0, stream>>>(
      Hj_h, 0, H, Wpj_h, 0, 768, Zj, Zj_h, 0, 32, nullptr, 0, nullptr, (int)BS, 24, 768);
  gemm_h_kernel<<<dim3(1, 16, 1), T, 0, stream>>>(
      Hi_h, 0, H, Wpi_h, 0, 768, Zi, Zi_h, 0, 32, nullptr, 0, nullptr, (int)BS, 24, 768);

  // --- separable row terms: Arow = Zj@W1a^T + b1, Brow = Zi@W1b^T ---
  gemm_h_kernel<<<dim3(2, 16, 1), T, 0, stream>>>(
      Zj_h, 0, 32, W1a_h, 0, 32, Arw, nullptr, 0, MH, b1, 0, nullptr, (int)BS, MH, 32);
  gemm_h_kernel<<<dim3(2, 16, 1), T, 0, stream>>>(
      Zi_h, 0, 32, W1b_h, 0, 32, Brw, nullptr, 0, MH, nullptr, 0, nullptr, (int)BS, MH, 32);

  // --- fused pairwise MLP -> logits ---
  pair_logits_kernel<<<dim3(S / 16, S / 128, B), T, 0, stream>>>(
      Zj, Zi, Arw, Brw, W1cd_h, W2, b2, mask, lgbuf, S);

  // --- softmax over t ---
  softmax_kernel<<<(unsigned)BS, T, 0, stream>>>(lgbuf, probs_h, S);

  // --- ctx = probs @ H_i : batched, B-side = HiT_h [H][S] weight layout ---
  gemm_h_kernel<<<dim3(12, 4, B), T, 0, stream>>>(
      probs_h, (long)S * S, S, HiT_h, (long)H * S, S,
      ctx, nullptr, (long)S * H, H, nullptr, 0, nullptr, S, H, S);

  // --- msg_in = [ctx, H_j, ctx*H_j] ---
  msgin_kernel<<<g1(BS * H), T, 0, stream>>>(ctx, H_j, msgin, BS, H);

  // --- val MLP: hid2 = relu(msg_in@Wv1^T + bv1); out = alpha*(hid2@Wv2^T + bv2) ---
  gemm_h_kernel<<<dim3(12, 16, 1), T, 0, stream>>>(
      msgin, 0, 2304, Wv1_h, 0, 2304, nullptr, hid2, 0, H, bv1, 1, nullptr, (int)BS, H, 2304);
  gemm_h_kernel<<<dim3(12, 16, 1), T, 0, stream>>>(
      hid2, 0, H, Wv2_h, 0, 768, outF, nullptr, 0, H, bv2, 0, alpha, (int)BS, H, 768);
}
